// SAGENET_90761248899607
// MI455X (gfx1250) — compile-verified, matrix operand order checked by reading, not run on hardware
//
#include <hip/hip_runtime.h>
#include <stdint.h>

// ---------------- CDNA5 vector types ----------------
typedef __attribute__((ext_vector_type(16))) __bf16 v16bf;
typedef __attribute__((ext_vector_type(8)))  __bf16 v8bf;
typedef __attribute__((ext_vector_type(4)))  __bf16 v4bf;
typedef __attribute__((ext_vector_type(8)))  float  v8f;

#define NNODES 8192
#define INF    16384
#define H1F    4096
#define H2F    2048
#define H3F    1024
#define OUTF   10

// ---------------- helpers ----------------
// Pack two f32 -> two bf16 (round-half-up) in 3 VALU: 2x v_add + 1x v_perm_b32.
static __device__ __forceinline__ uint32_t pack2_bf16(float f0, float f1) {
  uint32_t u0 = __builtin_bit_cast(uint32_t, f0) + 0x8000u;
  uint32_t u1 = __builtin_bit_cast(uint32_t, f1) + 0x8000u;
  // dst = { u1[31:16], u0[31:16] } : v_perm_b32 with S0=u1 (bytes 7..4), S1=u0 (bytes 3..0)
  return __builtin_amdgcn_perm(u1, u0, 0x07060302u);
}

// Build one 16x32 bf16 WMMA fragment from a padded LDS row.
// ISA 7.12.2 (16-bit A 16x32): lane half 0 holds K {0..7,16..23}, half 1 holds
// K {8..15,24..31}; caller passes p = rowBase + half*8 so the four groups sit
// at element offsets {0,4,16,20} (8B-aligned; lowers to 2x ds_load_b128).
static __device__ __forceinline__ v16bf load_frag(const __bf16* __restrict__ p) {
  v4bf g0 = *(const v4bf*)(p +  0);
  v4bf g1 = *(const v4bf*)(p +  4);
  v4bf g2 = *(const v4bf*)(p + 16);
  v4bf g3 = *(const v4bf*)(p + 20);
  v8bf h0 = __builtin_shufflevector(g0, g1, 0,1,2,3,4,5,6,7);
  v8bf h1 = __builtin_shufflevector(g2, g3, 0,1,2,3,4,5,6,7);
  return __builtin_shufflevector(h0, h1, 0,1,2,3,4,5,6,7,8,9,10,11,12,13,14,15);
}

// ---------------- tiled bf16 WMMA GEMM (double-buffered, BK=64) ----------------
// D[M,N] = act( A[M,K] @ W[N,K]^T (+bias[N]) (+addend[M,N]) )
// BM=BN=128, BK=64. 256 threads = 8 waves; wave grid 4(M)x2(N); each wave owns
// 32x64 = 2x4 WMMA tiles and issues 16 WMMAs per K-step (two 32-wide substeps).
// M,N multiples of 128; K multiple of 64. LDS ping-pong: one barrier / K-step;
// next tile's global loads issue before the current tile's WMMAs.
template<bool RELU, bool BIAS, bool ADD>
__global__ __launch_bounds__(256) void gemm_bf16_wmma(
    const float* __restrict__ A, const float* __restrict__ W,
    const float* __restrict__ bias, const float* __restrict__ addend,
    float* __restrict__ D, int M, int N, int K)
{
  constexpr int BM = 128, BN = 128, BK = 64, LDSS = 72; // 72*2B = 144B rows (16B aligned)
  __shared__ __bf16 As[2][BM * LDSS];
  __shared__ __bf16 Bs[2][BN * LDSS];

  const int tid  = threadIdx.x;
  const int lane = tid & 31;
  const int wave = tid >> 5;
  const int wm   = wave >> 1;      // 0..3
  const int wn   = wave & 1;       // 0..1
  const int half = lane >> 4;      // 0,1
  const int l15  = lane & 15;
  const int blockM = blockIdx.y * BM;
  const int blockN = blockIdx.x * BN;

  // staging: thread -> (row 0..127, 32-float half-row)
  const int lrow  = tid >> 1;
  const int lhalf = tid & 1;
  const float* aRow = A + (size_t)(blockM + lrow) * K + lhalf * 32;
  const float* wRow = W + (size_t)(blockN + lrow) * K + lhalf * 32;
  const int sOff = lrow * LDSS + lhalf * 32;

  float4 ra[8], rb[8];
  auto load32 = [](const float* __restrict__ g, float4 r[8]) {
    const float4* p = (const float4*)g;
#pragma unroll
    for (int i = 0; i < 8; ++i) r[i] = p[i];
  };
  auto store32 = [](const float4 r[8], __bf16* __restrict__ s) {
#pragma unroll
    for (int i = 0; i < 4; ++i) {
      uint4 q;
      q.x = pack2_bf16(r[2*i  ].x, r[2*i  ].y);
      q.y = pack2_bf16(r[2*i  ].z, r[2*i  ].w);
      q.z = pack2_bf16(r[2*i+1].x, r[2*i+1].y);
      q.w = pack2_bf16(r[2*i+1].z, r[2*i+1].w);
      *(uint4*)(s + i * 8) = q;                 // ds_store_b128
    }
  };

  v8f acc[2][4];
#pragma unroll
  for (int i = 0; i < 2; ++i)
#pragma unroll
    for (int j = 0; j < 4; ++j)
      acc[i][j] = (v8f){0.f,0.f,0.f,0.f,0.f,0.f,0.f,0.f};

  // prologue: stage K-step 0 into buffer 0
  load32(aRow, ra);
  load32(wRow, rb);
  store32(ra, &As[0][sOff]);
  store32(rb, &Bs[0][sOff]);
  __syncthreads();

  int cur = 0;
  for (int k0 = 0; k0 < K; k0 += BK) {
    const bool hasNext = (k0 + BK) < K;
    // issue next tile's global loads first (overlap with WMMAs below)
    if (hasNext) {
      load32(aRow + (k0 + BK), ra);
      load32(wRow + (k0 + BK), rb);
      if (k0 + 2 * BK < K) {       // warm L2 two steps ahead (global_prefetch_b8)
        __builtin_prefetch(aRow + (k0 + 2 * BK), 0, 1);
        __builtin_prefetch(wRow + (k0 + 2 * BK), 0, 1);
      }
    }

    // compute current buffer: two 32-wide K substeps -> 16 WMMAs
    const __bf16* asC = As[cur];
    const __bf16* bsC = Bs[cur];
#pragma unroll
    for (int kk = 0; kk < BK; kk += 32) {
      v16bf af[2], bf[4];
#pragma unroll
      for (int mt = 0; mt < 2; ++mt)
        af[mt] = load_frag(asC + (wm * 32 + mt * 16 + l15) * LDSS + kk + half * 8);
#pragma unroll
      for (int nt = 0; nt < 4; ++nt)
        bf[nt] = load_frag(bsC + (wn * 64 + nt * 16 + l15) * LDSS + kk + half * 8);
#pragma unroll
      for (int mt = 0; mt < 2; ++mt)
#pragma unroll
        for (int nt = 0; nt < 4; ++nt)
          acc[mt][nt] = __builtin_amdgcn_wmma_f32_16x16x32_bf16(
              false, af[mt], false, bf[nt], (short)0, acc[mt][nt], false, false);
    }

    // stage next tile into the other buffer; the previous iteration's barrier
    // already fenced everyone's reads of that buffer.
    if (hasNext) {
      store32(ra, &As[cur ^ 1][sOff]);
      store32(rb, &Bs[cur ^ 1][sOff]);
    }
    __syncthreads();
    cur ^= 1;
  }

  // epilogue: C/D layout -> VGPR r holds (M = r + 8*half, N = l15)
#pragma unroll
  for (int mt = 0; mt < 2; ++mt) {
#pragma unroll
    for (int nt = 0; nt < 4; ++nt) {
      const int n = blockN + wn * 64 + nt * 16 + l15;
      float bv = BIAS ? bias[n] : 0.f;
#pragma unroll
      for (int r = 0; r < 8; ++r) {
        const int m = blockM + wm * 32 + mt * 16 + half * 8 + r;
        float v = acc[mt][nt][r] + bv;
        if (ADD)  v += addend[(size_t)m * N + n];
        if (RELU) v = fmaxf(v, 0.f);
        D[(size_t)m * N + n] = v;
      }
    }
  }
}

// ---------------- graph kernels ----------------
__global__ void zero_i32(int* __restrict__ p, int n) {
  int i = blockIdx.x * blockDim.x + threadIdx.x;
  if (i < n) p[i] = 0;
}

__global__ void count_deg(const long long* __restrict__ ei, int* __restrict__ cnt, int E) {
  int e = blockIdx.x * blockDim.x + threadIdx.x;
  if (e < E) {
    int dst = (int)ei[E + e];
    atomicAdd(&cnt[dst], 1);
  }
}

// exclusive scan of cnt[0..8191] -> row[0..8192], single 256-thread workgroup
__global__ __launch_bounds__(256) void scan_rows(const int* __restrict__ cnt,
                                                 int* __restrict__ row) {
  __shared__ int part[256];
  const int tid = threadIdx.x;
  const int base = tid * 32;
  int loc[32];
  int s = 0;
#pragma unroll
  for (int i = 0; i < 32; ++i) { loc[i] = s; s += cnt[base + i]; }
  part[tid] = s;
  __syncthreads();
  if (tid == 0) {
    int a = 0;
    for (int i = 0; i < 256; ++i) { int t = part[i]; part[i] = a; a += t; }
    row[NNODES] = a;
  }
  __syncthreads();
  const int off = part[tid];
#pragma unroll
  for (int i = 0; i < 32; ++i) row[base + i] = off + loc[i];
}

__global__ void inv_deg_k(const int* __restrict__ cnt, float* __restrict__ inv, int n) {
  int i = blockIdx.x * blockDim.x + threadIdx.x;
  if (i < n) inv[i] = (cnt[i] > 0) ? (1.0f / (float)cnt[i]) : 0.0f;
}

__global__ void fill_csr(const long long* __restrict__ ei, const int* __restrict__ row,
                         int* __restrict__ cursor, int* __restrict__ csr_src, int E) {
  int e = blockIdx.x * blockDim.x + threadIdx.x;
  if (e < E) {
    int dst = (int)ei[E + e];
    int src = (int)ei[e];
    int pos = atomicAdd(&cursor[dst], 1);
    csr_src[row[dst] + pos] = src;
  }
}

// mean-aggregate incoming neighbors: block per node, features striped over 256 threads
__global__ __launch_bounds__(256) void aggregate(const float* __restrict__ h,
                                                 const int* __restrict__ row,
                                                 const int* __restrict__ csr_src,
                                                 const float* __restrict__ inv,
                                                 float* __restrict__ agg, int F) {
  const int node = blockIdx.x;
  const int beg = row[node], end = row[node + 1];
  const float s = inv[node];
  const int fpt = F >> 8;          // 8 (F=2048) or 4 (F=1024)
  float acc[8];
#pragma unroll
  for (int i = 0; i < 8; ++i) acc[i] = 0.f;
  for (int e = beg; e < end; ++e) {
    const float* hp = h + (size_t)csr_src[e] * F + threadIdx.x;
#pragma unroll 8
    for (int i = 0; i < fpt; ++i) acc[i] += hp[i * 256];
  }
  float* op = agg + (size_t)node * F + threadIdx.x;
#pragma unroll 8
  for (int i = 0; i < fpt; ++i) op[i * 256] = acc[i] * s;
}

// ---------------- final SAGE layer (1024 -> 10) + ReLU + softmax ----------------
// one wave per node; lanes stripe K, shuffle-reduce 10 partials
__global__ __launch_bounds__(256) void sage2_softmax(
    const float* __restrict__ agg, const float* __restrict__ h,
    const float* __restrict__ lw, const float* __restrict__ lb,
    const float* __restrict__ rw, float* __restrict__ out)
{
  const int node = blockIdx.x * 8 + (threadIdx.x >> 5);
  const int lane = threadIdx.x & 31;
  float acc[OUTF];
#pragma unroll
  for (int n = 0; n < OUTF; ++n) acc[n] = 0.f;
  const float* ag = agg + (size_t)node * H3F;
  const float* hh = h   + (size_t)node * H3F;
  for (int k = lane; k < H3F; k += 32) {
    float av = ag[k], hv = hh[k];
#pragma unroll
    for (int n = 0; n < OUTF; ++n)
      acc[n] += av * lw[n * H3F + k] + hv * rw[n * H3F + k];
  }
#pragma unroll
  for (int n = 0; n < OUTF; ++n) {
#pragma unroll
    for (int off = 16; off >= 1; off >>= 1)
      acc[n] += __shfl_xor(acc[n], off, 32);
  }
  if (lane == 0) {
    float v[OUTF], mx = -1e30f;
#pragma unroll
    for (int n = 0; n < OUTF; ++n) { v[n] = fmaxf(acc[n] + lb[n], 0.f); mx = fmaxf(mx, v[n]); }
    float s = 0.f;
#pragma unroll
    for (int n = 0; n < OUTF; ++n) { v[n] = __expf(v[n] - mx); s += v[n]; }
    float is = 1.f / s;
#pragma unroll
    for (int n = 0; n < OUTF; ++n) out[(size_t)node * OUTF + n] = v[n] * is;
  }
}

// ---------------- workspace layout ----------------
static constexpr size_t SZ_BUFA  = (size_t)NNODES * H1F * 4;   // 134 MB: h1, then agg1|P1|h3
static constexpr size_t SZ_BUFB  = (size_t)NNODES * H2F * 4;   // 67 MB: h2, then agg2
static constexpr size_t OFF_BUFA = 0;
static constexpr size_t OFF_BUFB = OFF_BUFA + SZ_BUFA;
static constexpr size_t OFF_CNT  = OFF_BUFB + SZ_BUFB;         // int[8192]
static constexpr size_t OFF_ROW  = OFF_CNT + 8192 * 4;         // int[8193] (padded)
static constexpr size_t OFF_CUR  = OFF_ROW + 33024;            // int[8192]
static constexpr size_t OFF_INV  = OFF_CUR + 8192 * 4;         // float[8192]
static constexpr size_t OFF_CSR  = OFF_INV + 8192 * 4;         // int[E]

extern "C" void kernel_launch(void* const* d_in, const int* in_sizes, int n_in,
                              void* d_out, int out_size, void* d_ws, size_t ws_size,
                              hipStream_t stream) {
  const float*      x    = (const float*)d_in[0];
  const long long*  ei   = (const long long*)d_in[1];   // int64 edge_index [2,E]
  const float*      w1   = (const float*)d_in[2];
  const float*      b1   = (const float*)d_in[3];
  const float*      w2   = (const float*)d_in[4];
  const float*      b2   = (const float*)d_in[5];
  const float*      s1lw = (const float*)d_in[6];
  const float*      s1lb = (const float*)d_in[7];
  const float*      s1rw = (const float*)d_in[8];
  const float*      s2lw = (const float*)d_in[9];
  const float*      s2lb = (const float*)d_in[10];
  const float*      s2rw = (const float*)d_in[11];
  float*            out  = (float*)d_out;

  const int E = in_sizes[1] / 2;
  char* ws = (char*)d_ws;
  if (ws_size < OFF_CSR + (size_t)E * 4) return;   // insufficient scratch

  float* bufA   = (float*)(ws + OFF_BUFA);
  float* bufB   = (float*)(ws + OFF_BUFB);
  int*   cnt    = (int*)(ws + OFF_CNT);
  int*   row    = (int*)(ws + OFF_ROW);
  int*   cursor = (int*)(ws + OFF_CUR);
  float* invd   = (float*)(ws + OFF_INV);
  int*   csr    = (int*)(ws + OFF_CSR);

  float* h1   = bufA;                                   // [8192,4096]
  float* h2   = bufB;                                   // [8192,2048]
  float* agg1 = bufA;                                   // [8192,2048] (reuses h1)
  float* P1   = bufA + (size_t)NNODES * H2F;            // [8192,1024]
  float* h3   = P1   + (size_t)NNODES * H3F;            // [8192,1024]
  float* agg2 = bufB;                                   // [8192,1024] (reuses h2)

  // ---- CSR build (cheap; independent of GEMM1) ----
  zero_i32<<<(NNODES + 255) / 256, 256, 0, stream>>>(cnt, NNODES);
  zero_i32<<<(NNODES + 255) / 256, 256, 0, stream>>>(cursor, NNODES);
  count_deg<<<(E + 255) / 256, 256, 0, stream>>>(ei, cnt, E);
  scan_rows<<<1, 256, 0, stream>>>(cnt, row);
  inv_deg_k<<<(NNODES + 255) / 256, 256, 0, stream>>>(cnt, invd, NNODES);
  fill_csr<<<(E + 255) / 256, 256, 0, stream>>>(ei, row, cursor, csr, E);

  // ---- h1 = relu(x @ w1^T + b1) : 8192x4096, K=16384 ----
  gemm_bf16_wmma<true, true, false>
      <<<dim3(H1F / 128, NNODES / 128), 256, 0, stream>>>(
          x, w1, b1, nullptr, h1, NNODES, H1F, INF);

  // ---- h2 = relu(h1 @ w2^T + b2) : 8192x2048, K=4096 ----
  gemm_bf16_wmma<true, true, false>
      <<<dim3(H2F / 128, NNODES / 128), 256, 0, stream>>>(
          h1, w2, b2, nullptr, h2, NNODES, H2F, H1F);

  // ---- SAGE1: P1 = h2 @ s1_r^T ; agg1 = mean-agg(h2) ; h3 = relu(agg1 @ s1_l^T + b + P1)
  gemm_bf16_wmma<false, false, false>
      <<<dim3(H3F / 128, NNODES / 128), 256, 0, stream>>>(
          h2, s1rw, nullptr, nullptr, P1, NNODES, H3F, H2F);
  aggregate<<<NNODES, 256, 0, stream>>>(h2, row, csr, invd, agg1, H2F);
  gemm_bf16_wmma<true, true, true>
      <<<dim3(H3F / 128, NNODES / 128), 256, 0, stream>>>(
          agg1, s1lw, s1lb, P1, h3, NNODES, H3F, H2F);

  // ---- SAGE2 + softmax ----
  aggregate<<<NNODES, 256, 0, stream>>>(h3, row, csr, invd, agg2, H3F);
  sage2_softmax<<<NNODES / 8, 256, 0, stream>>>(agg2, h3, s2lw, s2lb, s2rw, out);
}
